// LineEGCNII_1374389534972
// MI455X (gfx1250) — compile-verified, matrix operand order checked by reading, not run on hardware
//
#include <hip/hip_runtime.h>
#include <math.h>

typedef __attribute__((ext_vector_type(16))) _Float16 v16h;
typedef __attribute__((ext_vector_type(8)))  float    v8f;

#define N_NODES   20000
#define N_EDGES   160000     // line-graph node count
#define N_LEDGES  1280000    // line-graph edge count
#define IN_FEAT   128
#define HIDDEN    64
#define FEAT      128        // 2*HIDDEN
#define OUT_FEAT  16
#define NUM_LAYERS 8

// ---------------- WMMA fragment helpers (16-bit A/B layout per CDNA5 ISA) ---
// A (16x32, MxK): lane%16 = M; lane/16 selects K-half; v[0..7]=K(kb..kb+7),
// v[8..15]=K(kb+16..kb+23), kb = kk*32 + (lane/16)*8.
__device__ __forceinline__ v16h a_frag_lds(const _Float16* As, int lane, int kk) {
  int m  = lane & 15;
  int kb = kk * 32 + ((lane >> 4) << 3);
  const _Float16* rp = As + m * 128;
  v16h a;
#pragma unroll
  for (int j = 0; j < 8; ++j) {
    a[j]     = rp[kb + j];
    a[j + 8] = rp[kb + 16 + j];
  }
  return a;
}
// B (32x16, KxN): lane%16 = N; same K striping. Wt is pre-transposed: Wt[n][k].
__device__ __forceinline__ v16h b_frag_glb(const _Float16* Wt, int n, int lane, int kk) {
  int kb = kk * 32 + ((lane >> 4) << 3);
  const _Float16* rp = Wt + n * 128;
  v16h b;
#pragma unroll
  for (int j = 0; j < 8; ++j) {
    b[j]     = rp[kb + j];
    b[j + 8] = rp[kb + 16 + j];
  }
  return b;
}

// ---------------- weight prep: transpose + f32->f16 ------------------------
__global__ void k_prep_w(const float* __restrict__ w0, const float* __restrict__ cw,
                         const float* __restrict__ w1, _Float16* __restrict__ W0t,
                         _Float16* __restrict__ Wt, _Float16* __restrict__ W1t) {
  int idx = blockIdx.x * blockDim.x + threadIdx.x;
  if (idx < 64 * 128) {                     // lin0_w [128,64] -> W0t[64][128]
    int n = idx >> 7, k = idx & 127;
    W0t[idx] = (_Float16)w0[k * 64 + n];
  } else if (idx < 64 * 128 + 8 * 128 * 128) { // conv_w [8,128,128] -> Wt[l][128][128]
    int t = idx - 64 * 128;
    int l = t >> 14, rem = t & 16383;
    int n = rem >> 7, k = rem & 127;
    Wt[t] = (_Float16)cw[l * 16384 + k * 128 + n];
  } else if (idx < 64 * 128 + 8 * 128 * 128 + 16 * 128) { // lin1_w [128,16] -> W1t[16][128]
    int t = idx - (64 * 128 + 8 * 128 * 128);
    int n = t >> 7, k = t & 127;
    W1t[t] = (_Float16)w1[k * 16 + n];
  }
}

// ---------------- h = relu(x @ W0 + b0) : WMMA, 4 col-tiles per block -------
__global__ void __launch_bounds__(128) k_lin0(const float* __restrict__ x,
        const _Float16* __restrict__ W0t, const float* __restrict__ b0,
        float* __restrict__ h) {
  __shared__ _Float16 As[16 * 128];
  int tid = threadIdx.x;
  int rt  = blockIdx.x;                       // 0..1249, 16 rows each
  const float* xp = x + (size_t)rt * 16 * IN_FEAT;
  for (int i = tid; i < 16 * 128; i += 128) As[i] = (_Float16)xp[i];
  __syncthreads();
  int ct = tid >> 5, lane = tid & 31;         // one wave per 16-col tile
  v8f c = {};
#pragma unroll
  for (int kk = 0; kk < 4; ++kk) {
    v16h a = a_frag_lds(As, lane, kk);
    v16h b = b_frag_glb(W0t, ct * 16 + (lane & 15), lane, kk);
    c = __builtin_amdgcn_wmma_f32_16x16x32_f16(false, a, false, b, (short)0, c, false, false);
  }
  int n = ct * 16 + (lane & 15);
  float bias = b0[n];
#pragma unroll
  for (int v = 0; v < 8; ++v) {
    int m = v + ((lane >> 4) << 3);
    float r = c[v] + bias;
    h[(size_t)(rt * 16 + m) * HIDDEN + n] = r > 0.f ? r : 0.f;
  }
}

// ---------------- e0 = [h[src] || h[dst]] ; also init E = X0 ----------------
__global__ void k_concat(const float* __restrict__ h, const int* __restrict__ src,
                         const int* __restrict__ dst, float* __restrict__ X0,
                         float* __restrict__ E) {
  int idx = blockIdx.x * blockDim.x + threadIdx.x;  // < N_EDGES*128
  int e = idx >> 7, f = idx & 127;
  int node = (f < 64) ? src[e] : dst[e];
  float v = h[node * 64 + (f & 63)];
  X0[idx] = v;
  E[idx]  = v;
}

// ---------------- degree / norm / CSR build ---------------------------------
__global__ void k_deg_init(int* deg, int* rc) {
  int i = blockIdx.x * blockDim.x + threadIdx.x;
  if (i < N_EDGES) { deg[i] = 1; rc[i] = 0; }   // deg starts at 1 (self loop)
}
__global__ void k_deg_count(const int* __restrict__ row, const int* __restrict__ col,
                            int* deg, int* rc) {
  int i = blockIdx.x * blockDim.x + threadIdx.x;
  atomicAdd(&deg[col[i]], 1);
  atomicAdd(&rc[row[i]], 1);
}
__global__ void k_dis(const int* __restrict__ deg, float* __restrict__ dis) {
  int i = blockIdx.x * blockDim.x + threadIdx.x;
  if (i < N_EDGES) dis[i] = rsqrtf((float)deg[i]);
}
// single-block chunked exclusive scan of rc -> rp (and wr-offset copy wo)
__global__ void __launch_bounds__(1024) k_scan(const int* __restrict__ rc,
        int* __restrict__ rp, int* __restrict__ wo) {
  __shared__ int s[1024];
  int tid = threadIdx.x;
  const int CH = (N_EDGES + 1023) / 1024;       // 157
  int lo = tid * CH;
  int hi = lo + CH; if (hi > N_EDGES) hi = N_EDGES; if (lo > N_EDGES) lo = N_EDGES;
  int sum = 0;
  for (int i = lo; i < hi; ++i) sum += rc[i];
  s[tid] = sum;
  for (int off = 1; off < 1024; off <<= 1) {
    __syncthreads();
    int t = (tid >= off) ? s[tid - off] : 0;
    __syncthreads();
    s[tid] += t;
  }
  __syncthreads();
  int run = s[tid] - sum;                        // exclusive prefix
  for (int i = lo; i < hi; ++i) { rp[i] = run; wo[i] = run; run += rc[i]; }
  if (tid == 1023) rp[N_EDGES] = run;
}
__global__ void k_scatter(const int* __restrict__ row, const int* __restrict__ col,
        const float* __restrict__ dis, int* wo, int* __restrict__ ccol,
        float* __restrict__ cval) {
  int i = blockIdx.x * blockDim.x + threadIdx.x;
  int r = row[i], c = col[i];
  int pos = atomicAdd(&wo[r], 1);
  ccol[pos] = c;
  cval[pos] = dis[r] * dis[c];
}

// ---------------- SpMM: m = 0.9*(A_hat e) + 0.1*x0 ; 32 lanes x float4/row --
__global__ void __launch_bounds__(256) k_spmm(const float* __restrict__ E,
        const float* __restrict__ X0, const int* __restrict__ rp,
        const int* __restrict__ ccol, const float* __restrict__ cval,
        const float* __restrict__ dis, float* __restrict__ M) {
  int tid = threadIdx.x;
  int lane = tid & 31;
  int r = blockIdx.x * 8 + (tid >> 5);
  const float4* ep = (const float4*)E;
  float sv = dis[r] * dis[r];                    // self-loop weight 1/deg
  float4 a = ep[(size_t)r * 32 + lane];
  float ax = sv * a.x, ay = sv * a.y, az = sv * a.z, aw = sv * a.w;
  int j0 = rp[r], j1 = rp[r + 1];
  for (int j = j0; j < j1; ++j) {
    int c = ccol[j]; float v = cval[j];
    float4 w = ep[(size_t)c * 32 + lane];
    ax += v * w.x; ay += v * w.y; az += v * w.z; aw += v * w.w;
  }
  const float4* xp = (const float4*)X0;
  float4 x0 = xp[(size_t)r * 32 + lane];
  float4 o;
  o.x = 0.9f * ax + 0.1f * x0.x;
  o.y = 0.9f * ay + 0.1f * x0.y;
  o.z = 0.9f * az + 0.1f * x0.z;
  o.w = 0.9f * aw + 0.1f * x0.w;
  ((float4*)M)[(size_t)r * 32 + lane] = o;
}

// ---------------- e = relu((1-b)*m + b*(m@W)) : WMMA, 8 col-tiles/block -----
__global__ void __launch_bounds__(256) k_gcn2_gemm(const float* __restrict__ M,
        const _Float16* __restrict__ Wt, float beta, float* __restrict__ E) {
  __shared__ _Float16 Ah[16 * 128];
  __shared__ float    Af[16 * 128];
  int tid = threadIdx.x;
  int rt  = blockIdx.x;                          // 0..9999
  const float* mp = M + (size_t)rt * 16 * 128;
  for (int i = tid; i < 16 * 128; i += 256) {
    float v = mp[i]; Af[i] = v; Ah[i] = (_Float16)v;
  }
  __syncthreads();
  int ct = tid >> 5, lane = tid & 31;
  v8f c = {};
#pragma unroll
  for (int kk = 0; kk < 4; ++kk) {
    v16h a = a_frag_lds(Ah, lane, kk);
    v16h b = b_frag_glb(Wt, ct * 16 + (lane & 15), lane, kk);
    c = __builtin_amdgcn_wmma_f32_16x16x32_f16(false, a, false, b, (short)0, c, false, false);
  }
  int n = ct * 16 + (lane & 15);
  float omb = 1.0f - beta;
#pragma unroll
  for (int v = 0; v < 8; ++v) {
    int m = v + ((lane >> 4) << 3);
    float r = omb * Af[m * 128 + n] + beta * c[v]; // identity term in f32
    E[(size_t)(rt * 16 + m) * 128 + n] = r > 0.f ? r : 0.f;
  }
}

// ---------------- out = e @ W1 + b1 : one wave per row-tile, 1 col-tile -----
__global__ void __launch_bounds__(256) k_final(const float* __restrict__ E,
        const _Float16* __restrict__ W1t, const float* __restrict__ b1,
        float* __restrict__ out) {
  int tid = threadIdx.x;
  int wave = tid >> 5, lane = tid & 31;
  int rt = blockIdx.x * 8 + wave;                // 0..9999
  const float* rp = E + (size_t)(rt * 16 + (lane & 15)) * 128;
  v8f c = {};
#pragma unroll
  for (int kk = 0; kk < 4; ++kk) {
    int kb = kk * 32 + ((lane >> 4) << 3);
    v16h a;
#pragma unroll
    for (int j = 0; j < 8; ++j) {
      a[j]     = (_Float16)rp[kb + j];
      a[j + 8] = (_Float16)rp[kb + 16 + j];
    }
    v16h b = b_frag_glb(W1t, lane & 15, lane, kk);
    c = __builtin_amdgcn_wmma_f32_16x16x32_f16(false, a, false, b, (short)0, c, false, false);
  }
  int n = lane & 15;
  float bias = b1[n];
#pragma unroll
  for (int v = 0; v < 8; ++v) {
    int m = v + ((lane >> 4) << 3);
    out[(size_t)(rt * 16 + m) * OUT_FEAT + n] = c[v] + bias;
  }
}

// ---------------------------------------------------------------------------
extern "C" void kernel_launch(void* const* d_in, const int* in_sizes, int n_in,
                              void* d_out, int out_size, void* d_ws, size_t ws_size,
                              hipStream_t stream) {
  const float* x  = (const float*)d_in[0];
  const int*   ei = (const int*)d_in[1];
  const int*   lei= (const int*)d_in[2];
  const float* w0 = (const float*)d_in[3];
  const float* b0 = (const float*)d_in[4];
  const float* cw = (const float*)d_in[5];
  const float* w1 = (const float*)d_in[6];
  const float* b1 = (const float*)d_in[7];
  float* out = (float*)d_out;

  char* p = (char*)d_ws;
  auto alloc = [&](size_t bytes) -> void* {
    void* r = (void*)p;
    p += (bytes + 255) & ~(size_t)255;
    return r;
  };
  float*    H    = (float*)alloc((size_t)N_NODES * HIDDEN * 4);
  float*    X0   = (float*)alloc((size_t)N_EDGES * FEAT * 4);
  float*    E    = (float*)alloc((size_t)N_EDGES * FEAT * 4);
  float*    M    = (float*)alloc((size_t)N_EDGES * FEAT * 4);
  int*      deg  = (int*)alloc((size_t)N_EDGES * 4);
  float*    dis  = (float*)alloc((size_t)N_EDGES * 4);
  int*      rc   = (int*)alloc((size_t)N_EDGES * 4);
  int*      rp   = (int*)alloc((size_t)(N_EDGES + 1) * 4);
  int*      wo   = (int*)alloc((size_t)N_EDGES * 4);
  int*      ccol = (int*)alloc((size_t)N_LEDGES * 4);
  float*    cval = (float*)alloc((size_t)N_LEDGES * 4);
  _Float16* W0t  = (_Float16*)alloc(64 * 128 * 2);
  _Float16* Wt   = (_Float16*)alloc(8 * 128 * 128 * 2);
  _Float16* W1t  = (_Float16*)alloc(16 * 128 * 2);

  const int* src  = ei;
  const int* dst  = ei + N_EDGES;
  const int* lrow = lei;
  const int* lcol = lei + N_LEDGES;

  k_prep_w   <<<552, 256, 0, stream>>>(w0, cw, w1, W0t, Wt, W1t);
  k_lin0     <<<N_NODES / 16, 128, 0, stream>>>(x, W0t, b0, H);
  k_concat   <<<N_EDGES * FEAT / 256, 256, 0, stream>>>(H, src, dst, X0, E);
  k_deg_init <<<(N_EDGES + 255) / 256, 256, 0, stream>>>(deg, rc);
  k_deg_count<<<N_LEDGES / 256, 256, 0, stream>>>(lrow, lcol, deg, rc);
  k_dis      <<<(N_EDGES + 255) / 256, 256, 0, stream>>>(deg, dis);
  k_scan     <<<1, 1024, 0, stream>>>(rc, rp, wo);
  k_scatter  <<<N_LEDGES / 256, 256, 0, stream>>>(lrow, lcol, dis, wo, ccol, cval);

  for (int l = 0; l < NUM_LAYERS; ++l) {
    float beta = logf(0.5f / (float)(l + 1) + 1.0f);
    k_spmm     <<<N_EDGES / 8, 256, 0, stream>>>(E, X0, rp, ccol, cval, dis, M);
    k_gcn2_gemm<<<N_EDGES / 16, 256, 0, stream>>>(M, Wt + (size_t)l * 16384, beta, E);
  }
  k_final<<<N_EDGES / 16 / 8, 256, 0, stream>>>(E, W1t, b1, out);
}
